// GNN_68650757259640
// MI455X (gfx1250) — matrix operand order repaired and checked by reading, not compiled, not run on hardware
//
#include <hip/hip_runtime.h>
#include <hip/hip_bf16.h>
#include <math.h>

#define N_NODES 100000
#define N_EDGES 1600000
#define E2      (2 * N_EDGES)
#define NTILES  (E2 / 16)          // 200000 exactly, no remainder
#define HN      10                 // node/edge feature width
#define HID     64

typedef __attribute__((ext_vector_type(16))) _Float16 v16h;
typedef __attribute__((ext_vector_type(2)))  _Float16 v2h;
typedef __attribute__((ext_vector_type(8)))  float    v8f;

union UPK  { unsigned u; _Float16 f[2]; };
union UV16 { v16h v; unsigned u[8]; };

static __device__ __forceinline__ v8f wmma16(v16h A, v16h B, v8f C) {
    // D(16x16 f32) = A(16x32 f16) x B(32x16 f16) + C
    return __builtin_amdgcn_wmma_f32_16x16x32_f16(
        false, A, false, B, (short)0, C, false, false);
}

// pack 2xf32 -> 2xf16 (one v_cvt_pk_f16_f32), then relu in f16
// (one v_pk_max_num_f16).  relu(cvt(x)) == cvt(relu(x)) exactly.
static __device__ __forceinline__ unsigned pack_relu2(float a, float b) {
    UPK p;
    p.f[0] = (_Float16)a;
    p.f[1] = (_Float16)b;
    v2h x    = __builtin_bit_cast(v2h, p.u);
    v2h zero = {(_Float16)0.f, (_Float16)0.f};
    v2h r    = __builtin_elementwise_max(x, zero);
    return __builtin_bit_cast(unsigned, r);
}

// ---------------------------------------------------------------------------
// Transposed edge MLP:  m^T = W3 relu(W2 relu(W1 [h_s;h_d;1]^T))  per edge,
// edges on the WMMA N-side (lane = edge column), weights register-resident
// on the A-side. Inter-layer D->B relayout = one half-wave shfl_xor per dword.
// Then a[src] += m (global atomics).  Zero LDS.
// ---------------------------------------------------------------------------
__global__ __launch_bounds__(256) void edge_mlp_kernel(
    const float* __restrict__ h, const int* __restrict__ edges,
    const float* __restrict__ W1, const float* __restrict__ b1,
    const float* __restrict__ W2, const float* __restrict__ b2,
    const float* __restrict__ W3, const float* __restrict__ b3,
    float* __restrict__ a)
{
    const int lane = threadIdx.x & 31;
    const int wv   = threadIdx.x >> 5;
    const int hi   = lane >> 4;          // half-wave: 0 = lanes 0-15
    const int ln   = lane & 15;

    // ---- A fragments (contiguous-k A layout used here: m = lane&15,
    //      half h -> k = h + 16*hi), loaded once ----
    // W1': [64 x 32], k<20 = W1, k==20 = b1 (bias column), else 0.
    v16h AW1[4];
    #pragma unroll
    for (int mt = 0; mt < 4; ++mt) {
        const int m = 16 * mt + ln;
        #pragma unroll
        for (int w = 0; w < 8; ++w) {
            const int kb = 2 * w + 16 * hi;
            #pragma unroll
            for (int e = 0; e < 2; ++e) {
                const int k = kb + e;
                float v = (k < 2 * HN) ? W1[m * (2 * HN) + k]
                                       : ((k == 2 * HN) ? b1[m] : 0.f);
                AW1[mt][2 * w + e] = (_Float16)v;
            }
        }
    }
    v16h AW2[4][2];
    #pragma unroll
    for (int mt = 0; mt < 4; ++mt) {
        const int m = 16 * mt + ln;
        #pragma unroll
        for (int ks = 0; ks < 2; ++ks)
            #pragma unroll
            for (int w = 0; w < 8; ++w) {
                const int kb = 32 * ks + 2 * w + 16 * hi;
                #pragma unroll
                for (int e = 0; e < 2; ++e)
                    AW2[mt][ks][2 * w + e] = (_Float16)W2[m * HID + kb + e];
            }
    }
    v16h AW3[2];   // rows 10..15 zero-padded
    #pragma unroll
    for (int ks = 0; ks < 2; ++ks)
        #pragma unroll
        for (int w = 0; w < 8; ++w) {
            const int kb = 32 * ks + 2 * w + 16 * hi;
            #pragma unroll
            for (int e = 0; e < 2; ++e)
                AW3[ks][2 * w + e] = (ln < HN) ? (_Float16)W3[ln * HID + kb + e]
                                               : (_Float16)0.f;
        }

    // ---- biases as WMMA C-operands (D row = 16*mt + 8*hi + r) ----
    v8f cb2[4];
    #pragma unroll
    for (int mt = 0; mt < 4; ++mt)
        #pragma unroll
        for (int r = 0; r < 8; ++r)
            cb2[mt][r] = b2[16 * mt + 8 * hi + r];
    v8f cb3;
    #pragma unroll
    for (int r = 0; r < 8; ++r) {
        const int f = r + 8 * hi;
        cb3[r] = (f < HN) ? b3[f] : 0.f;
    }

    const int gw     = blockIdx.x * 8 + wv;
    const int stride = gridDim.x * 8;

    for (int t = gw; t < NTILES; t += stride) {
        // ---- edge for this lane's column (both half-waves: same edge) ----
        const int e    = t * 16 + ln;                 // < E2 (exact)
        const int j    = (e < N_EDGES) ? e : e - N_EDGES;
        const int flip = (e >= N_EDGES);
        const int s    = edges[flip ? (N_EDGES + j) : j];
        const int d    = edges[flip ? j : (N_EDGES + j)];
        const float* hs = h + s * HN;
        const float* hd = h + d * HN;

        // ---- x^T B-fragment: lane = edge, k = h + 16*hi ----
        //  k 0..9 = h[src], 10..19 = h[dst], 20 = 1.0 (bias), 21..31 = 0
        v16h Bx;
        if (hi == 0) {
            #pragma unroll
            for (int q = 0; q < HN; ++q) Bx[q] = (_Float16)hs[q];
            #pragma unroll
            for (int q = 0; q < 6; ++q)  Bx[HN + q] = (_Float16)hd[q];
        } else {
            #pragma unroll
            for (int q = 0; q < 4; ++q)  Bx[q] = (_Float16)hd[6 + q];
            Bx[4] = (_Float16)1.0f;
            #pragma unroll
            for (int q = 5; q < 16; ++q) Bx[q] = (_Float16)0.f;
        }

        // ---- layer 1: D1 = W1' x x'  (bias folded into K=20 column) ----
        v8f c[4];
        #pragma unroll
        for (int mt = 0; mt < 4; ++mt) {
            v8f z = {0.f, 0.f, 0.f, 0.f, 0.f, 0.f, 0.f, 0.f};
            c[mt] = wmma16(AW1[mt], Bx, z);
        }

        // ---- pack f16 + relu -> D->B relayout via half-wave swap ----
        unsigned pk[4][4];
        #pragma unroll
        for (int mt = 0; mt < 4; ++mt)
            #pragma unroll
            for (int w = 0; w < 4; ++w)
                pk[mt][w] = pack_relu2(c[mt][2 * w], c[mt][2 * w + 1]);
        v16h Bm[2];
        #pragma unroll
        for (int ks = 0; ks < 2; ++ks) {
            UV16 bb;
            #pragma unroll
            for (int w = 0; w < 4; ++w) {
                unsigned keep = hi ? pk[2 * ks + 1][w] : pk[2 * ks][w];
                unsigned send = hi ? pk[2 * ks][w]     : pk[2 * ks + 1][w];
                unsigned got  = __shfl_xor(send, 16, 32);
                bb.u[w]     = hi ? got  : keep;
                bb.u[4 + w] = hi ? keep : got;
            }
            Bm[ks] = bb.v;
        }

        // ---- layer 2: D2 = W2 x m1 + b2 (bias via C operand) ----
        #pragma unroll
        for (int mt = 0; mt < 4; ++mt) {
            c[mt] = wmma16(AW2[mt][0], Bm[0], cb2[mt]);
            c[mt] = wmma16(AW2[mt][1], Bm[1], c[mt]);
        }
        #pragma unroll
        for (int mt = 0; mt < 4; ++mt)
            #pragma unroll
            for (int w = 0; w < 4; ++w)
                pk[mt][w] = pack_relu2(c[mt][2 * w], c[mt][2 * w + 1]);
        #pragma unroll
        for (int ks = 0; ks < 2; ++ks) {
            UV16 bb;
            #pragma unroll
            for (int w = 0; w < 4; ++w) {
                unsigned keep = hi ? pk[2 * ks + 1][w] : pk[2 * ks][w];
                unsigned send = hi ? pk[2 * ks][w]     : pk[2 * ks + 1][w];
                unsigned got  = __shfl_xor(send, 16, 32);
                bb.u[w]     = hi ? got  : keep;
                bb.u[4 + w] = hi ? keep : got;
            }
            Bm[ks] = bb.v;
        }

        // ---- layer 3: D3 = W3 x m2 + b3 ----
        v8f c3 = wmma16(AW3[0], Bm[0], cb3);
        c3     = wmma16(AW3[1], Bm[1], c3);

        // ---- scatter: a[src][f] += m  (f = r + 8*hi; valid f < 10) ----
        float* ap = a + s * HN;
        if (hi == 0) {
            #pragma unroll
            for (int r = 0; r < 8; ++r) atomicAdd(ap + r, c3[r]);
        } else {
            atomicAdd(ap + 8, c3[0]);
            atomicAdd(ap + 9, c3[1]);
        }
    }
}

// ---------------------------------------------------------------------------
// Per-node single-step GRU (torch gate math).
// ---------------------------------------------------------------------------
__global__ __launch_bounds__(256) void gru_kernel(
    const float* __restrict__ a, const float* __restrict__ hin,
    const float* __restrict__ Wih, const float* __restrict__ bih,
    const float* __restrict__ Whh, const float* __restrict__ bhh,
    float* __restrict__ hout)
{
    __shared__ float sWih[30][10], sWhh[30][10], sbih[30], sbhh[30];
    int tid = threadIdx.x;
    for (int i = tid; i < 300; i += blockDim.x) {
        sWih[i / 10][i % 10] = Wih[i];
        sWhh[i / 10][i % 10] = Whh[i];
    }
    if (tid < 30) { sbih[tid] = bih[tid]; sbhh[tid] = bhh[tid]; }
    __syncthreads();

    int n = blockIdx.x * blockDim.x + tid;
    if (n >= N_NODES) return;

    float av[10], hv[10];
    #pragma unroll
    for (int q = 0; q < 10; ++q) { av[q] = a[n * 10 + q]; hv[q] = hin[n * 10 + q]; }

    #pragma unroll
    for (int k = 0; k < 10; ++k) {
        float ir = sbih[k],       hr = sbhh[k];
        float iz = sbih[10 + k],  hz = sbhh[10 + k];
        float in_ = sbih[20 + k], hn = sbhh[20 + k];
        #pragma unroll
        for (int q = 0; q < 10; ++q) {
            ir  += sWih[k][q] * av[q];       hr += sWhh[k][q] * hv[q];
            iz  += sWih[10 + k][q] * av[q];  hz += sWhh[10 + k][q] * hv[q];
            in_ += sWih[20 + k][q] * av[q];  hn += sWhh[20 + k][q] * hv[q];
        }
        float r  = 1.f / (1.f + expf(-(ir + hr)));
        float z  = 1.f / (1.f + expf(-(iz + hz)));
        float nn = tanhf(in_ + r * hn);
        hout[n * 10 + k] = (1.f - z) * nn + z * hv[k];
    }
}

__global__ __launch_bounds__(256) void zero_kernel(float* __restrict__ p, int n) {
    int i = blockIdx.x * blockDim.x + threadIdx.x;
    if (i < n) p[i] = 0.f;
}

// ---------------------------------------------------------------------------
extern "C" void kernel_launch(void* const* d_in, const int* in_sizes, int n_in,
                              void* d_out, int out_size, void* d_ws, size_t ws_size,
                              hipStream_t stream)
{
    const float* node_features = (const float*)d_in[0];
    const int*   edges         = (const int*)d_in[1];
    const float* W1 = (const float*)d_in[2];   const float* b1 = (const float*)d_in[3];
    const float* W2 = (const float*)d_in[4];   const float* b2 = (const float*)d_in[5];
    const float* W3 = (const float*)d_in[6];   const float* b3 = (const float*)d_in[7];
    const float* Wih = (const float*)d_in[8];  const float* bih = (const float*)d_in[9];
    const float* Whh = (const float*)d_in[10]; const float* bhh = (const float*)d_in[11];
    float* out = (float*)d_out;

    const int NF = N_NODES * HN;               // 1,000,000
    float* a  = (float*)d_ws;                  // [N,10] segment-sum accumulator
    float* hA = a + NF;
    float* hB = hA + NF;

    const float* hcur = node_features;
    float* houts[3] = { hA, hB, out };

    for (int it = 0; it < 3; ++it) {
        zero_kernel<<<(NF + 255) / 256, 256, 0, stream>>>(a, NF);
        edge_mlp_kernel<<<1024, 256, 0, stream>>>(hcur, edges, W1, b1, W2, b2, W3, b3, a);
        gru_kernel<<<(N_NODES + 255) / 256, 256, 0, stream>>>(a, hcur, Wih, bih, Whh, bhh, houts[it]);
        hcur = houts[it];
    }
}
